// BayesianBertSelfAttention_11123965296977
// MI455X (gfx1250) — matrix operand order repaired and checked by reading, not compiled, not run on hardware
//
#include <hip/hip_runtime.h>
#include <hip/hip_bf16.h>

#define B_    2
#define S_    1024
#define HID_  768
#define NH_   12
#define HD_   64
#define MAXP_ 1024
#define EROWS 2048   // dist_emb rows padded 2047 -> 2048

typedef __attribute__((ext_vector_type(16))) __bf16 v16bf;
typedef __attribute__((ext_vector_type(8)))  float  v8f;

union F16x16 { v16bf v; uint4 q[2]; };
union BF4    { __bf16 e[4]; uint2 u; };

__device__ __forceinline__ v16bf ld_frag_2x16B(const void* p0, const void* p1) {
  F16x16 f;
  f.q[0] = *(const uint4*)p0;
  f.q[1] = *(const uint4*)p1;
  return f.v;
}
__device__ __forceinline__ v16bf ld_frag_32B(const void* p) {
  F16x16 f;
  const uint4* q = (const uint4*)p;
  f.q[0] = q[0];
  f.q[1] = q[1];
  return f.v;
}
__device__ __forceinline__ v8f wmma_bf16(v16bf a, v16bf b, v8f c) {
  return __builtin_amdgcn_wmma_f32_16x16x32_bf16(false, a, false, b, (short)0, c,
                                                 false, false);
}
// reductions across the 16 lanes sharing one C-matrix row
__device__ __forceinline__ float rowmax16(float x) {
  x = fmaxf(x, __shfl_xor(x, 1));
  x = fmaxf(x, __shfl_xor(x, 2));
  x = fmaxf(x, __shfl_xor(x, 4));
  x = fmaxf(x, __shfl_xor(x, 8));
  return x;
}
__device__ __forceinline__ float rowsum16(float x) {
  x += __shfl_xor(x, 1);
  x += __shfl_xor(x, 2);
  x += __shfl_xor(x, 4);
  x += __shfl_xor(x, 8);
  return x;
}

// One async-to-LDS b128 op: per-lane 16B, per-wave 512B.  INST_OFFSET adds to
// both the LDS (VDST) and global (SADDR+VADDR) addresses.
__device__ __forceinline__ void async_b128(unsigned lds, unsigned voff,
                                           const void* saddr) {
  asm volatile("global_load_async_to_lds_b128 %0, %1, %2 offset:0"
               :: "v"(lds), "v"(voff), "s"(saddr) : "memory");
}
// Contiguous 4KB tile, block = 128 threads: 2 b128 ops per wave.
__device__ __forceinline__ void async_stage_4k(unsigned lds_base,
                                               const __bf16* g, int tid) {
  unsigned o = (unsigned)tid * 16u;
  asm volatile(
      "global_load_async_to_lds_b128 %0, %1, %2 offset:0\n\t"
      "global_load_async_to_lds_b128 %0, %1, %2 offset:2048"
      :: "v"(lds_base + o), "v"(o), "s"(g)
      : "memory");
}
__device__ __forceinline__ void wait_asynccnt_le(int n) {
  asm volatile("s_wait_asynccnt %0" :: "i"(n) : "memory");
}

// ------------------------------------------------------------ pre-passes
__global__ void cvt_h_kernel(const float* __restrict__ in,
                             __bf16* __restrict__ out) {   // 4 elems/thread
  int i = blockIdx.x * blockDim.x + threadIdx.x;
  float4 f = ((const float4*)in)[i];
  BF4 o;
  o.e[0] = (__bf16)f.x; o.e[1] = (__bf16)f.y;
  o.e[2] = (__bf16)f.z; o.e[3] = (__bf16)f.w;
  ((uint2*)out)[i] = o.u;
}

__global__ void cvt_dist_kernel(const float* __restrict__ dist,
                                __bf16* __restrict__ Ebf) {  // over EROWS*64/4
  int i = blockIdx.x * blockDim.x + threadIdx.x;
  int j = i >> 4;                       // row (16 chunks of 4 per 64-wide row)
  int d4 = (i & 15) * 4;
  int js = (j < 2 * MAXP_ - 1) ? j : (2 * MAXP_ - 2);
  float4 f = *(const float4*)&dist[js * HD_ + d4];
  BF4 o;
  o.e[0] = (__bf16)f.x; o.e[1] = (__bf16)f.y;
  o.e[2] = (__bf16)f.z; o.e[3] = (__bf16)f.w;
  ((uint2*)Ebf)[i] = o.u;
}

// W[k][n] f32 -> Wt[z][n][k] bf16 (transposed, async-stageable)
__global__ void cvt_w_kernel(const float* __restrict__ Wq,
                             const float* __restrict__ Wk,
                             const float* __restrict__ Wv,
                             __bf16* __restrict__ Wt) {
  int z = blockIdx.y;
  const float* W = (z == 0) ? Wq : (z == 1) ? Wk : Wv;
  int i = blockIdx.x * blockDim.x + threadIdx.x;   // over 768*192
  int n  = i / (HID_ / 4);
  int k4 = (i % (HID_ / 4)) * 4;
  BF4 o;
#pragma unroll
  for (int j = 0; j < 4; ++j) o.e[j] = (__bf16)W[(size_t)(k4 + j) * HID_ + n];
  ((uint2*)(Wt + (size_t)z * HID_ * HID_))[i] = o.u;
}

// ---------------------------------------------------------------- kernel 1
// Q/K/V projection.  Block = 128 threads (4 waves), block tile 64x64, each
// wave a 32x32 tile (4 WMMAs per K-step).  A/B tiles DMA'd into rotating
// triple-buffered LDS via async-to-LDS; one barrier per K-step.
__global__ void qkv_kernel(const __bf16* __restrict__ Hbf,   // [2048][768]
                           const __bf16* __restrict__ Wt,    // [3][768n][768k]
                           const float* __restrict__ bq, const float* __restrict__ bk,
                           const float* __restrict__ bv,
                           __bf16* __restrict__ Qbf, __bf16* __restrict__ Kbf,
                           __bf16* __restrict__ Vbf) {
  const int z = blockIdx.z;
  const float* bias = (z == 0) ? bq : (z == 1) ? bk : bv;
  __bf16* Out       = (z == 0) ? Qbf : (z == 1) ? Kbf : Vbf;
  const __bf16* Wz  = Wt + (size_t)z * HID_ * HID_;

  const int m0 = blockIdx.x * 64, n0 = blockIdx.y * 64;
  const int tid = threadIdx.x, lane = tid & 31, wave = tid >> 5;
  const int moff = (wave & 1) * 32, noff = (wave >> 1) * 32;
  const int lrow = lane & 15;
  const int ak0 = (lane < 16) ? 0 : 16;
  const int bk0 = (lane < 16) ? 0 : 16;

  __shared__ __align__(16) __bf16 As[3][64][32];   // [m][k]
  __shared__ __align__(16) __bf16 Bs[3][64][32];   // [n][k]

  // staging map (128 threads): tile = 256 x 16B chunks, 2 per thread
  const unsigned crow = (unsigned)(tid >> 2), part = (unsigned)(tid & 3);
  const unsigned ldso = crow * 64u + part * 16u;            // chunk2: +2048
  const unsigned avo0 = (unsigned)((m0 + crow) * HID_) * 2u + part * 16u;
  const unsigned avo1 = avo0 + 32u * HID_ * 2u;
  const unsigned bvo0 = (unsigned)((n0 + crow) * HID_) * 2u + part * 16u;
  const unsigned bvo1 = bvo0 + 32u * HID_ * 2u;

  __bf16 *A0 = &As[0][0][0], *A1 = &As[1][0][0], *A2 = &As[2][0][0];
  __bf16 *B0 = &Bs[0][0][0], *B1 = &Bs[1][0][0], *B2 = &Bs[2][0][0];

  // prologue: tile kk=0 into current buffers (4 ops/wave)
  async_b128((unsigned)(uintptr_t)A0 + ldso,         avo0, Hbf);
  async_b128((unsigned)(uintptr_t)A0 + ldso + 2048u, avo1, Hbf);
  async_b128((unsigned)(uintptr_t)B0 + ldso,         bvo0, Wz);
  async_b128((unsigned)(uintptr_t)B0 + ldso + 2048u, bvo1, Wz);

  v8f acc[2][2] = {{v8f{}, v8f{}}, {v8f{}, v8f{}}};
  for (int kk = 0; kk < HID_; kk += 32) {
    if (kk + 32 < HID_) {
      async_b128((unsigned)(uintptr_t)A1 + ldso,         avo0, Hbf + kk + 32);
      async_b128((unsigned)(uintptr_t)A1 + ldso + 2048u, avo1, Hbf + kk + 32);
      async_b128((unsigned)(uintptr_t)B1 + ldso,         bvo0, Wz + kk + 32);
      async_b128((unsigned)(uintptr_t)B1 + ldso + 2048u, bvo1, Wz + kk + 32);
      wait_asynccnt_le(4);   // in-order: current tile's 4 ops complete
    } else {
      wait_asynccnt_le(0);
    }
    __syncthreads();         // all waves' chunks visible

    v16bf a[2], b[2];
#pragma unroll
    for (int mt = 0; mt < 2; ++mt) {
      const __bf16* ar = A0 + (moff + mt * 16 + lrow) * 32;
      a[mt] = ld_frag_2x16B(ar + ak0, ar + ak0 + 8);
    }
#pragma unroll
    for (int nt = 0; nt < 2; ++nt)
      b[nt] = ld_frag_32B(B0 + (noff + nt * 16 + lrow) * 32 + bk0);
#pragma unroll
    for (int mt = 0; mt < 2; ++mt)
#pragma unroll
      for (int nt = 0; nt < 2; ++nt)
        acc[mt][nt] = wmma_bf16(a[mt], b[nt], acc[mt][nt]);

    // rotate buffers (no modulo arithmetic, no trailing barrier needed)
    __bf16* t;
    t = A0; A0 = A1; A1 = A2; A2 = t;
    t = B0; B0 = B1; B1 = B2; B2 = t;
  }

#pragma unroll
  for (int mt = 0; mt < 2; ++mt)
#pragma unroll
  for (int nt = 0; nt < 2; ++nt) {
    int n = n0 + noff + nt * 16 + lrow;
    int h = n >> 6, d = n & 63;
    float bval = bias[n];
#pragma unroll
    for (int v = 0; v < 8; ++v) {
      int m = m0 + moff + mt * 16 + ((lane < 16) ? v : v + 8);
      int bi = m >> 10, s = m & 1023;
      Out[((size_t)(bi * NH_ + h) * S_ + s) * HD_ + d] =
          (__bf16)(acc[mt][nt][v] + bval);
    }
  }
}

// ---------------------------------------------------------------- kernel 2
// Fused dual-branch flash attention with relative-position bias.
// K/V tiles rotate through triple-buffered LDS (async DMA); Vt flip-flops.
__global__ void attn_kernel(const __bf16* __restrict__ Qbf,
                            const __bf16* __restrict__ Kbf,
                            const __bf16* __restrict__ Vbf,
                            const __bf16* __restrict__ Ebf,
                            const float* __restrict__ amask,   // [B,S]
                            const float* __restrict__ smask,   // [S,S]
                            const float* __restrict__ sel,     // [B,S]
                            float* __restrict__ out) {         // [B,S,HID]
  const int tid  = threadIdx.x;
  const int lane = tid & 31;
  const int wave = tid >> 5;
  const int bh = blockIdx.y;
  const int b = bh / NH_, h = bh % NH_;
  const int l0 = blockIdx.x * 64 + wave * 16;
  const int lrow = lane & 15;
  const bool lohalf = (lane < 16);
  const int ak0 = lohalf ? 0 : 16;
  const int bk0 = lohalf ? 0 : 16;

  const size_t headoff = (size_t)(b * NH_ + h) * S_ * HD_;
  const __bf16* Qh = Qbf + headoff;
  const __bf16* Kh = Kbf + headoff;
  const __bf16* Vh = Vbf + headoff;

  __shared__ __align__(16) __bf16 Ks[3][32][64];  // K tiles, triple-buffered
  __shared__ __align__(16) __bf16 Vs[3][32][64];  // V tiles (row-major)
  __shared__ __align__(16) __bf16 Vt[2][64][32];  // V transposed, double-buf
  __shared__ __align__(16) float  Pb[4][16][33];  // per-wave rel-bias scratch
  __shared__ __align__(16) __bf16 Pp[4][16][32];  // per-wave probs relayout

  __bf16 *K0 = &Ks[0][0][0], *K1 = &Ks[1][0][0], *K2 = &Ks[2][0][0];
  __bf16 *V0 = &Vs[0][0][0], *V1 = &Vs[1][0][0], *V2 = &Vs[2][0][0];
  __bf16 *T0 = &Vt[0][0][0], *T1 = &Vt[1][0][0];

  // persistent Q fragments
  const __bf16* qrow = Qh + (size_t)(l0 + lrow) * HD_;
  v16bf aq0 = ld_frag_2x16B(qrow + ak0, qrow + ak0 + 8);
  v16bf aq1 = ld_frag_2x16B(qrow + 32 + ak0, qrow + 32 + ak0 + 8);

  float rm[2][8], rs[2][8];
  v8f ctx[2][4];
#pragma unroll
  for (int br = 0; br < 2; ++br) {
#pragma unroll
    for (int v = 0; v < 8; ++v) { rm[br][v] = -1e30f; rs[br][v] = 0.f; }
#pragma unroll
    for (int dt = 0; dt < 4; ++dt) ctx[br][dt] = v8f{};
  }
  float sS[2][2][8];

  // prologue: stage tile 0 (4 ops/wave)
  async_stage_4k((unsigned)(uintptr_t)K0, Kh, tid);
  async_stage_4k((unsigned)(uintptr_t)V0, Vh, tid);

  for (int r0 = 0; r0 < S_; r0 += 32) {
    if (r0 + 32 < S_) {
      async_stage_4k((unsigned)(uintptr_t)K1, Kh + (size_t)(r0 + 32) * HD_, tid);
      async_stage_4k((unsigned)(uintptr_t)V1, Vh + (size_t)(r0 + 32) * HD_, tid);
      wait_asynccnt_le(4);
    } else {
      wait_asynccnt_le(0);
    }
    __syncthreads();   // current tile visible in K0/V0

    // vectorized LDS->LDS transpose V0 -> T0
#pragma unroll
    for (int e = 0; e < 2; ++e) {
      int c = e * 128 + tid;          // 256 chunks of 8 bf16
      int r = c >> 3, dp = (c & 7) * 8;
      uint4 q = *(const uint4*)(V0 + r * 64 + dp);
      const __bf16* vv = (const __bf16*)&q;
#pragma unroll
      for (int j = 0; j < 8; ++j) T0[(dp + j) * 32 + r] = vv[j];
    }
    __syncthreads();   // T0 visible

#pragma unroll
    for (int t = 0; t < 2; ++t) {
      const int rsub = r0 + t * 16;
      const int j0 = l0 - rsub + (MAXP_ - 16);   // >= 0, j0+31 <= 2047
#pragma unroll
      for (int pt = 0; pt < 2; ++pt) {
        const __bf16* erow = Ebf + (size_t)(j0 + pt * 16 + lrow) * HD_;
        v16bf be0 = ld_frag_32B(erow + bk0);
        v16bf be1 = ld_frag_32B(erow + 32 + bk0);
        v8f pb = v8f{};
        pb = wmma_bf16(aq0, be0, pb);
        pb = wmma_bf16(aq1, be1, pb);
#pragma unroll
        for (int v = 0; v < 8; ++v)
          Pb[wave][lohalf ? v : v + 8][pt * 16 + lrow] = pb[v];
      }
      const __bf16* krow = K0 + (t * 16 + lrow) * 64;
      v16bf bkf0 = ld_frag_32B(krow + bk0);
      v16bf bkf1 = ld_frag_32B(krow + 32 + bk0);
      v8f qk = v8f{};
      qk = wmma_bf16(aq0, bkf0, qk);
      qk = wmma_bf16(aq1, bkf1, qk);

      const float am = amask[b * S_ + rsub + lrow];
#pragma unroll
      for (int v = 0; v < 8; ++v) {
        int rr = lohalf ? v : v + 8;
        float bias = Pb[wave][rr][rr - lrow + 15];
        float sg = (qk[v] + bias) * 0.125f + am;
        float sl = sg * smask[(size_t)(l0 + rr) * S_ + rsub + lrow] + am;
        sS[0][t][v] = sg;
        sS[1][t][v] = sl;
      }
    }

    v16bf bvf[4];
#pragma unroll
    for (int dt = 0; dt < 4; ++dt)
      bvf[dt] = ld_frag_32B(T0 + (dt * 16 + lrow) * 32 + bk0);

#pragma unroll
    for (int br = 0; br < 2; ++br) {
#pragma unroll
      for (int v = 0; v < 8; ++v) {
        int rr = lohalf ? v : v + 8;
        float tmax = rowmax16(fmaxf(sS[br][0][v], sS[br][1][v]));
        float mnew = fmaxf(rm[br][v], tmax);
        float corr = __expf(rm[br][v] - mnew);
        float p0 = __expf(sS[br][0][v] - mnew);
        float p1 = __expf(sS[br][1][v] - mnew);
        float psum = rowsum16(p0 + p1);
        rs[br][v] = rs[br][v] * corr + psum;
        rm[br][v] = mnew;
#pragma unroll
        for (int dt = 0; dt < 4; ++dt) ctx[br][dt][v] *= corr;
        Pp[wave][rr][lrow]      = (__bf16)p0;
        Pp[wave][rr][16 + lrow] = (__bf16)p1;
      }
      const __bf16* prow = &Pp[wave][lrow][0];
      v16bf ap = ld_frag_2x16B(prow + ak0, prow + ak0 + 8);
#pragma unroll
      for (int dt = 0; dt < 4; ++dt)
        ctx[br][dt] = wmma_bf16(ap, bvf[dt], ctx[br][dt]);
    }

    // rotate buffers (register moves only)
    __bf16* tp;
    tp = K0; K0 = K1; K1 = K2; K2 = tp;
    tp = V0; V0 = V1; V1 = V2; V2 = tp;
    tp = T0; T0 = T1; T1 = tp;
  }

#pragma unroll
  for (int v = 0; v < 8; ++v) {
    int rr = lohalf ? v : v + 8;
    int l = l0 + rr;
    float se = sel[b * S_ + l];
    float ig = 1.f / rs[0][v];
    float il = 1.f / rs[1][v];
#pragma unroll
    for (int dt = 0; dt < 4; ++dt) {
      int d = dt * 16 + lrow;
      float og = ctx[0][dt][v] * ig;
      float ol = ctx[1][dt][v] * il;
      out[(size_t)(b * S_ + l) * HID_ + h * HD_ + d] = se * ol + (1.f - se) * og;
    }
  }
}

// ---------------------------------------------------------------- launch
extern "C" void kernel_launch(void* const* d_in, const int* in_sizes, int n_in,
                              void* d_out, int out_size, void* d_ws, size_t ws_size,
                              hipStream_t stream) {
  const float* hid   = (const float*)d_in[0];
  const float* amask = (const float*)d_in[1];
  const float* smask = (const float*)d_in[2];
  const float* sel   = (const float*)d_in[3];
  const float* Wq    = (const float*)d_in[4];
  const float* bq    = (const float*)d_in[5];
  const float* Wk    = (const float*)d_in[6];
  const float* bk    = (const float*)d_in[7];
  const float* Wv    = (const float*)d_in[8];
  const float* bv    = (const float*)d_in[9];
  const float* dist  = (const float*)d_in[10];
  float* out = (float*)d_out;

  const size_t QKV = (size_t)B_ * NH_ * S_ * HD_;        // 1.5M elems each
  __bf16* Qbf = (__bf16*)d_ws;
  __bf16* Kbf = Qbf + QKV;
  __bf16* Vbf = Kbf + QKV;
  __bf16* Ebf = Vbf + QKV;                               // EROWS*HD
  __bf16* Hbf = Ebf + (size_t)EROWS * HD_;               // B*S*HID
  __bf16* Wt  = Hbf + (size_t)B_ * S_ * HID_;            // 3*HID*HID

  cvt_dist_kernel<<<(EROWS * HD_ / 4) / 256, 256, 0, stream>>>(dist, Ebf);
  cvt_h_kernel<<<(B_ * S_ * HID_ / 4) / 256, 256, 0, stream>>>(hid, Hbf);
  cvt_w_kernel<<<dim3((HID_ * HID_ / 4) / 256, 3), 256, 0, stream>>>(Wq, Wk, Wv, Wt);
  qkv_kernel<<<dim3(B_ * S_ / 64, HID_ / 64, 3), 128, 0, stream>>>(
      Hbf, Wt, bq, bk, bv, Qbf, Kbf, Vbf);
  attn_kernel<<<dim3(S_ / 64, B_ * NH_), 128, 0, stream>>>(
      Qbf, Kbf, Vbf, Ebf, amask, smask, sel, out);
}
